// SimplifiedMambaBlock_82523501625834
// MI455X (gfx1250) — matrix operand order repaired and checked by reading, not compiled, not run on hardware
//
#include <hip/hip_runtime.h>
#include <math.h>

#define B_SZ 16
#define SEQ  4096
#define DM   256
#define RTOT (B_SZ * SEQ)   // 65536 rows, kept in (s,b) order: r = s*16 + b
#define NE   768            // combined in-proj width: [x1(256) | x2(256) | u(256)]

typedef __attribute__((ext_vector_type(16))) __bf16 v16bf;
typedef __attribute__((ext_vector_type(8)))  __bf16 v8bf;
typedef __attribute__((ext_vector_type(8)))  float  v8f;

__device__ __forceinline__ v8f wmma_bf16(v16bf a, v16bf b, v8f c) {
  // D = A(16x32 bf16) x B(32x16 bf16) + C(16x16 f32)
  return __builtin_amdgcn_wmma_f32_16x16x32_bf16(false, a, false, b,
                                                 (short)0, c, false, false);
}

// A fragment (16x32 bf16) from row-major storage: rows 0..15 at `base`, cols k0..k0+31.
// Layout used here (same k order as load_b): lane&15 = row M; lanes 0-15 hold K = k0..k0+15 (contiguous),
// lanes 16-31 hold K = k0+16..k0+31.
__device__ __forceinline__ v16bf load_a(const __bf16* base, int lda, int k0, int lane) {
  const int row = lane & 15;
  const int kb  = k0 + ((lane >> 4) << 4);
  const __bf16* p = base + row * lda + kb;
  v8bf lo = *(const v8bf*)(p);
  v8bf hi = *(const v8bf*)(p + 8);
  v16bf r;
#pragma unroll
  for (int i = 0; i < 8; ++i) { r[i] = lo[i]; r[8 + i] = hi[i]; }
  return r;
}

// B fragment (32x16 bf16). Operand element B(k,n) is stored K-major: bt[n*ldb + k].
// Layout used here (same k order as load_a): lane&15 = column N; lanes 0-15 hold K = k0..k0+15 (contiguous),
// lanes 16-31 hold K = k0+16..k0+31.
__device__ __forceinline__ v16bf load_b(const __bf16* bt, int ldb, int n0, int k0, int lane) {
  const int col = lane & 15;
  const int kb  = k0 + ((lane >> 4) << 4);
  return *(const v16bf*)(bt + (size_t)(n0 + col) * ldb + kb);
}

// ---------------------------------------------------------------------------
// Prep: build bf16 weight operands.
//   bt1[e][d] (768x256): e<512 -> in_proj_w[e][d]; e>=512 -> (W1^T A)[d][e-512]
//   Bmb/Cmb/Wob: bf16 row-major copies of Bm, Cm, out_proj_w (already K-major for our B op)
//   ubias[j] = in_proj_b[0:256] @ A[:,j]   (bias folded into u)
// ---------------------------------------------------------------------------
__global__ void prep_kernel(const float* __restrict__ in_proj_w, const float* __restrict__ in_proj_b,
                            const float* __restrict__ A, const float* __restrict__ Bm,
                            const float* __restrict__ Cm, const float* __restrict__ out_proj_w,
                            __bf16* bt1, __bf16* Bmb, __bf16* Cmb, __bf16* Wob, float* ubias) {
  const int idx = blockIdx.x * blockDim.x + threadIdx.x;
  if (idx < NE * DM) {
    const int e = idx >> 8, d = idx & 255;
    float v;
    if (e < 512) {
      v = in_proj_w[e * DM + d];
    } else {
      const int j = e - 512;
      float s = 0.f;
      for (int i = 0; i < DM; ++i) s += in_proj_w[i * DM + d] * A[i * DM + j];
      v = s;
    }
    bt1[idx] = (__bf16)v;
  }
  if (idx < DM * DM) {
    Bmb[idx] = (__bf16)Bm[idx];
    Cmb[idx] = (__bf16)Cm[idx];
    Wob[idx] = (__bf16)out_proj_w[idx];
  }
  if (idx < DM) {
    float s = 0.f;
    for (int i = 0; i < DM; ++i) s += in_proj_b[i] * A[i * DM + idx];
    ubias[idx] = s;
  }
}

// x (b,s,d) f32  ->  xb bf16 in (s,b,d) row order so 16-row M-tiles are contiguous.
__global__ void xcast_kernel(const float* __restrict__ x, __bf16* __restrict__ xb) {
  const size_t idx = (size_t)blockIdx.x * blockDim.x + threadIdx.x;  // RTOT*DM threads
  const int d = (int)(idx & 255);
  const size_t r = idx >> 8;
  const int s = (int)(r >> 4), b = (int)(r & 15);
  xb[idx] = (__bf16)x[((size_t)b * SEQ + s) * DM + d];
}

// ---------------------------------------------------------------------------
// GEMM1: [x1|x2|u] (65536 x 768) = xb (65536x256) @ bt1^T, f32 outputs.
// 1 WG per 16-row tile; 8 waves x 6 N-tiles each.
// ---------------------------------------------------------------------------
__global__ void __launch_bounds__(256) gemm_in_kernel(const __bf16* __restrict__ xb,
                                                      const __bf16* __restrict__ bt1,
                                                      const float* __restrict__ in_proj_b,
                                                      const float* __restrict__ ubias,
                                                      float* __restrict__ x1f,
                                                      float* __restrict__ x2f,
                                                      float* __restrict__ u) {
  const int lane = threadIdx.x & 31;
  const int wv   = threadIdx.x >> 5;            // 0..7
  const size_t r0 = (size_t)blockIdx.x * 16;
  const __bf16* abase = xb + r0 * DM;
  v8f acc[6];
#pragma unroll
  for (int j = 0; j < 6; ++j)
#pragma unroll
    for (int i = 0; i < 8; ++i) acc[j][i] = 0.f;

#pragma unroll
  for (int kc = 0; kc < 8; ++kc) {
    v16bf a = load_a(abase, DM, kc * 32, lane);
#pragma unroll
    for (int j = 0; j < 6; ++j) {
      v16bf b = load_b(bt1, DM, (wv * 6 + j) * 16, kc * 32, lane);
      acc[j] = wmma_bf16(a, b, acc[j]);
    }
  }
  const int ml = (lane >> 4) << 3;   // C/D layout: lanes 16-31 hold M = r+8
  const int nl = lane & 15;
#pragma unroll
  for (int j = 0; j < 6; ++j) {
    const int e0 = (wv * 6 + j) * 16;
#pragma unroll
    for (int i = 0; i < 8; ++i) {
      const size_t row = r0 + ml + i;
      const int e = e0 + nl;
      const float v = acc[j][i];
      if (e < 256)      x1f[row * DM + e]         = v + in_proj_b[e];
      else if (e < 512) x2f[row * DM + (e - 256)] = v + in_proj_b[e];
      else              u  [row * DM + (e - 512)] = v + ubias[e - 512];
    }
  }
}

// ---------------------------------------------------------------------------
// Scan (critical path): single WG, 16 waves; wave w owns columns [16w,16w+16).
// Bm fragments resident in VGPRs for all 4096 steps; h ping-pongs in LDS (bf16).
// Per step: 8 ds A-frag loads + 8 wmma + tanh + 1 barrier.
// ---------------------------------------------------------------------------
__global__ void __launch_bounds__(512) scan_kernel(const float* __restrict__ u,
                                                   const __bf16* __restrict__ Bmb,
                                                   __bf16* __restrict__ Hb) {
  const int lane = threadIdx.x & 31;
  const int wv   = threadIdx.x >> 5;            // 0..15, N-tile
  __shared__ __bf16 hbuf[2][16 * DM];           // 16 KB of 320 KB LDS

  v16bf bfr[8];
#pragma unroll
  for (int kc = 0; kc < 8; ++kc) bfr[kc] = load_b(Bmb, DM, wv * 16, kc * 32, lane);

  for (int i = threadIdx.x; i < 16 * DM; i += 512) {
    hbuf[0][i] = (__bf16)0.f;
    hbuf[1][i] = (__bf16)0.f;
  }
  __syncthreads();

  const int ml = (lane >> 4) << 3;
  const int nl = lane & 15;
  const int n  = wv * 16 + nl;

  for (int t = 0; t < SEQ; ++t) {
    const int cur = t & 1, nxt = cur ^ 1;
    const float* ub = u + (size_t)t * 16 * DM;
    if (t + 1 < SEQ)
      __builtin_prefetch(u + (size_t)(t + 1) * 16 * DM + (size_t)threadIdx.x * 8, 0, 0);

    v8f c;
#pragma unroll
    for (int i = 0; i < 8; ++i) c[i] = ub[(size_t)(ml + i) * DM + n];

#pragma unroll
    for (int kc = 0; kc < 8; ++kc) {
      v16bf a = load_a(&hbuf[cur][0], DM, kc * 32, lane);
      c = wmma_bf16(a, bfr[kc], c);
    }
#pragma unroll
    for (int i = 0; i < 8; ++i) {
      const float hv = tanhf(c[i]);
      const __bf16 hb = (__bf16)hv;
      hbuf[nxt][(ml + i) * DM + n] = hb;
      Hb[((size_t)t * 16 + ml + i) * DM + n] = hb;
    }
    __syncthreads();   // step t+1 reads hbuf[nxt], overwrites hbuf[cur]
  }
}

// ---------------------------------------------------------------------------
// GEMM2 + epilogue: yg = (H @ Cm^T + x1*Dv) * sigmoid(x2), bf16 out.
// ---------------------------------------------------------------------------
__global__ void __launch_bounds__(256) gemm_yc_kernel(const __bf16* __restrict__ Hb,
                                                      const __bf16* __restrict__ Cmb,
                                                      const float* __restrict__ x1f,
                                                      const float* __restrict__ x2f,
                                                      const float* __restrict__ Dv,
                                                      __bf16* __restrict__ ygb) {
  const int lane = threadIdx.x & 31;
  const int wv   = threadIdx.x >> 5;            // 0..7, 2 N-tiles each
  const size_t r0 = (size_t)blockIdx.x * 16;
  const __bf16* abase = Hb + r0 * DM;
  v8f acc[2];
#pragma unroll
  for (int j = 0; j < 2; ++j)
#pragma unroll
    for (int i = 0; i < 8; ++i) acc[j][i] = 0.f;

#pragma unroll
  for (int kc = 0; kc < 8; ++kc) {
    v16bf a = load_a(abase, DM, kc * 32, lane);
#pragma unroll
    for (int j = 0; j < 2; ++j) {
      v16bf b = load_b(Cmb, DM, (wv * 2 + j) * 16, kc * 32, lane);
      acc[j] = wmma_bf16(a, b, acc[j]);
    }
  }
  const int ml = (lane >> 4) << 3;
  const int nl = lane & 15;
#pragma unroll
  for (int j = 0; j < 2; ++j) {
    const int n0 = (wv * 2 + j) * 16;
#pragma unroll
    for (int i = 0; i < 8; ++i) {
      const size_t row = r0 + ml + i;
      const int nn = n0 + nl;
      float y = acc[j][i] + x1f[row * DM + nn] * Dv[nn];
      const float g = x2f[row * DM + nn];
      y *= 1.f / (1.f + __expf(-g));
      ygb[row * DM + nn] = (__bf16)y;
    }
  }
}

// ---------------------------------------------------------------------------
// GEMM3: out = yg @ out_proj_w^T + out_proj_b, written back to (b,s,e) order.
// ---------------------------------------------------------------------------
__global__ void __launch_bounds__(256) gemm_out_kernel(const __bf16* __restrict__ ygb,
                                                       const __bf16* __restrict__ Wob,
                                                       const float* __restrict__ out_proj_b,
                                                       float* __restrict__ out) {
  const int lane = threadIdx.x & 31;
  const int wv   = threadIdx.x >> 5;
  const size_t r0 = (size_t)blockIdx.x * 16;
  const __bf16* abase = ygb + r0 * DM;
  v8f acc[2];
#pragma unroll
  for (int j = 0; j < 2; ++j)
#pragma unroll
    for (int i = 0; i < 8; ++i) acc[j][i] = 0.f;

#pragma unroll
  for (int kc = 0; kc < 8; ++kc) {
    v16bf a = load_a(abase, DM, kc * 32, lane);
#pragma unroll
    for (int j = 0; j < 2; ++j) {
      v16bf b = load_b(Wob, DM, (wv * 2 + j) * 16, kc * 32, lane);
      acc[j] = wmma_bf16(a, b, acc[j]);
    }
  }
  const int ml = (lane >> 4) << 3;
  const int nl = lane & 15;
#pragma unroll
  for (int j = 0; j < 2; ++j) {
    const int n0 = (wv * 2 + j) * 16;
#pragma unroll
    for (int i = 0; i < 8; ++i) {
      const size_t row = r0 + ml + i;           // row = s*16 + b
      const int nn = n0 + nl;
      const int s = (int)(row >> 4), b = (int)(row & 15);
      out[((size_t)b * SEQ + s) * DM + nn] = acc[j][i] + out_proj_b[nn];
    }
  }
}

extern "C" void kernel_launch(void* const* d_in, const int* in_sizes, int n_in,
                              void* d_out, int out_size, void* d_ws, size_t ws_size,
                              hipStream_t stream) {
  const float* x          = (const float*)d_in[0];
  const float* in_proj_w  = (const float*)d_in[1];
  const float* in_proj_b  = (const float*)d_in[2];
  const float* A          = (const float*)d_in[3];
  const float* Bm         = (const float*)d_in[4];
  const float* Cm         = (const float*)d_in[5];
  const float* Dv         = (const float*)d_in[6];
  const float* out_proj_w = (const float*)d_in[7];
  const float* out_proj_b = (const float*)d_in[8];
  float* out = (float*)d_out;
  (void)in_sizes; (void)n_in; (void)out_size; (void)ws_size;

  // Workspace carve-out (~303 MB total)
  char* p = (char*)d_ws;
  auto take = [&](size_t bytes) -> char* {
    char* q = p;
    p += (bytes + 255) & ~(size_t)255;
    return q;
  };
  __bf16* bt1   = (__bf16*)take((size_t)NE * DM * 2);   // 768x256 bf16
  __bf16* Bmb   = (__bf16*)take((size_t)DM * DM * 2);
  __bf16* Cmb   = (__bf16*)take((size_t)DM * DM * 2);
  __bf16* Wob   = (__bf16*)take((size_t)DM * DM * 2);
  float*  ubias = (float*) take((size_t)DM * 4);
  __bf16* xb    = (__bf16*)take((size_t)RTOT * DM * 2); // 32 MB
  float*  x1f   = (float*) take((size_t)RTOT * DM * 4); // 64 MB
  float*  x2f   = (float*) take((size_t)RTOT * DM * 4); // 64 MB
  float*  u     = (float*) take((size_t)RTOT * DM * 4); // 64 MB
  __bf16* Hb    = (__bf16*)take((size_t)RTOT * DM * 2); // 32 MB
  __bf16* ygb   = (__bf16*)take((size_t)RTOT * DM * 2); // 32 MB

  prep_kernel<<<(NE * DM + 255) / 256, 256, 0, stream>>>(
      in_proj_w, in_proj_b, A, Bm, Cm, out_proj_w, bt1, Bmb, Cmb, Wob, ubias);

  xcast_kernel<<<(RTOT * DM) / 256, 256, 0, stream>>>(x, xb);

  gemm_in_kernel<<<RTOT / 16, 256, 0, stream>>>(xb, bt1, in_proj_b, ubias, x1f, x2f, u);

  scan_kernel<<<1, 512, 0, stream>>>(u, Bmb, Hb);

  gemm_yc_kernel<<<RTOT / 16, 256, 0, stream>>>(Hb, Cmb, x1f, x2f, Dv, ygb);

  gemm_out_kernel<<<RTOT / 16, 256, 0, stream>>>(ygb, Wob, out_proj_b, out);
}
